// CrossModalAttention_39685497815547
// MI455X (gfx1250) — matrix operand order repaired and checked by reading, not compiled, not run on hardware
//
#include <hip/hip_runtime.h>
#include <hip/hip_bf16.h>

typedef __attribute__((ext_vector_type(16))) _Float16 v16h;
typedef __attribute__((ext_vector_type(8)))  _Float16 v8h;
typedef __attribute__((ext_vector_type(8)))  float    v8f;
typedef __attribute__((ext_vector_type(4)))  float    v4f;

#define SEQ    2048
#define NVIS   2048
#define NHEAD  16
#define BATCH  2

#if defined(__gfx1250__)
#define USE_ASYNC_LDS 1
#endif

static __device__ __forceinline__ v16h pack16(v8h lo, v8h hi) {
  v16h r;
#pragma unroll
  for (int i = 0; i < 8; ++i) { r[i] = lo[i]; r[i + 8] = hi[i]; }
  return r;
}

// ---------------------------------------------------------------------------
// Weight transpose + f32->f16 convert:  Wt[n][k] = (f16) W[k][n]
// ---------------------------------------------------------------------------
__global__ void cma_wT_kernel(const float* __restrict__ W,
                              _Float16* __restrict__ Wt, int K, int N) {
  int idx = blockIdx.x * blockDim.x + threadIdx.x;
  if (idx >= K * N) return;
  int n = idx / K, k = idx - n * K;
  Wt[idx] = (_Float16)W[(size_t)k * N + n];
}

// ---------------------------------------------------------------------------
// Input projection GEMM:  O[M][512] = f16( X[M][1024] @ WtT + bias )
// Wt is [512][1024] (pre-transposed).  One wave -> 16x64 output strip.
// ---------------------------------------------------------------------------
__global__ void cma_proj_kernel(const float* __restrict__ X,
                                const _Float16* __restrict__ Wt,
                                const float* __restrict__ bias,
                                _Float16* __restrict__ O) {
  const int lane = threadIdx.x & 31;
  const int wid  = (blockIdx.x * blockDim.x + threadIdx.x) >> 5;
  const int ln = lane & 15, half = lane >> 4;
  const int nt = wid & 7;      // 512 / 64 strips
  const int mt = wid >> 3;
  const int mbase = mt * 16, nbase = nt * 64;

  v8f acc[4] = {};
  const float* xrow = X + (size_t)(mbase + ln) * 1024;

  for (int kk = 0; kk < 1024; kk += 32) {
    if (kk + 64 < 1024) __builtin_prefetch(xrow + kk + 64, 0, 0);
    // A tile: lane = row, elements 0..15 -> k = half*16+e
    v4f x0 = *(const v4f*)(xrow + kk + half * 16);
    v4f x1 = *(const v4f*)(xrow + kk + half * 16 + 4);
    v4f x2 = *(const v4f*)(xrow + kk + half * 16 + 8);
    v4f x3 = *(const v4f*)(xrow + kk + half * 16 + 12);
    v16h a;
#pragma unroll
    for (int i = 0; i < 4; ++i) {
      a[i]      = (_Float16)x0[i];
      a[4 + i]  = (_Float16)x1[i];
      a[8 + i]  = (_Float16)x2[i];
      a[12 + i] = (_Float16)x3[i];
    }
#pragma unroll
    for (int t = 0; t < 4; ++t) {
      // B tile: lane = column N, contiguous K-run half*16..+15 = row of Wt
      v16h bm = *(const v16h*)(Wt + (size_t)(nbase + t * 16 + ln) * 1024 + kk + half * 16);
      acc[t] = __builtin_amdgcn_wmma_f32_16x16x32_f16(
          false, a, false, bm, (short)0, acc[t], false, false);
    }
  }
#pragma unroll
  for (int t = 0; t < 4; ++t) {
    int col = nbase + t * 16 + ln;
    float bv = bias[col];
#pragma unroll
    for (int v = 0; v < 8; ++v) {
      int row = mbase + half * 8 + v;   // C/D: lanes 0-15 -> M=v, lanes 16-31 -> M=8+v
      O[(size_t)row * 512 + col] = (_Float16)(acc[t][v] + bv);
    }
  }
}

// ---------------------------------------------------------------------------
// V transpose:  Vt[((b*16+h)*32+dh)][nv] = Vrow[(b*NVIS+nv)][h*32+dh]
// ---------------------------------------------------------------------------
__global__ void cma_vT_kernel(const _Float16* __restrict__ Vrow,
                              _Float16* __restrict__ Vt) {
  int idx = blockIdx.x * blockDim.x + threadIdx.x;   // 2^21 total
  int nv = idx & (NVIS - 1);
  int dh = (idx >> 11) & 31;
  int h  = (idx >> 16) & 15;
  int b  = idx >> 20;
  Vt[idx] = Vrow[((size_t)(b * NVIS + nv)) * 512 + h * 32 + dh];
}

// ---------------------------------------------------------------------------
// Flash attention, block-cooperative:
//   block = 256 threads = 8 waves, all on the same (b,h); each wave owns one
//   16-query tile.  Per 32-key step, the whole block stages the shared
//   32x32 K panel and 32x32 V^T panel into LDS (double buffered, async
//   global->LDS copy on gfx1250, tracked by ASYNCcnt), so K/V L2 traffic
//   drops 8x and each wave's WMMA B operands come from wide LDS reads.
// ---------------------------------------------------------------------------
__global__ __launch_bounds__(256) void cma_attn_kernel(
    const _Float16* __restrict__ Q,    // [B*SEQ][512]
    const _Float16* __restrict__ K,    // [B*NVIS][512]
    const _Float16* __restrict__ Vt,   // [B*H*32][NVIS]
    _Float16* __restrict__ Att) {      // [B*SEQ][512]
  __shared__ __align__(64) _Float16 stageK[2][32 * 32];   // 2 x 2KB
  __shared__ __align__(64) _Float16 stageV[2][32 * 32];   // 2 x 2KB
  __shared__ __align__(64) _Float16 ldsP[8 * 16 * 32];    // per-wave P patch

  const int tid  = threadIdx.x;
  const int lane = tid & 31;
  const int wib  = tid >> 5;
  const int ln = lane & 15, half = lane >> 4;

  const int bh = blockIdx.x >> 4;          // 32 (b,h) groups x 16 blocks
  const int qg = blockIdx.x & 15;
  const int b  = bh >> 4, h = bh & 15;
  const int qbase = (qg * 8 + wib) * 16;   // this wave's 16 queries
  _Float16* P = ldsP + wib * 512;

  // --- staging-fill assignment: threads 0..127 -> K panel, 128..255 -> V panel
  const int  fr  = (tid >> 2) & 31;        // panel row
  const int  fc  = tid & 3;                // 16B chunk within row
  const bool isV = tid >= 128;
  const _Float16* gbase = isV
      ? Vt + ((size_t)((b * NHEAD + h) * 32 + fr)) * NVIS + fc * 8
      : K  + ((size_t)(b * NVIS + fr)) * 512 + h * 32 + fc * 8;
  _Float16* lbase = (isV ? &stageV[0][0] : &stageK[0][0]) + fr * 32 + fc * 8;

  auto fillKV = [&](int buf, int kb) {
    const _Float16* g = gbase + (isV ? (size_t)kb : (size_t)kb * 512);
    _Float16* l = lbase + buf * 1024;
#if defined(USE_ASYNC_LDS)
    // VDST = per-lane LDS byte offset (low 32 bits of flat shared pointer),
    // VADDR = 64-bit global address.  Tracked by ASYNCcnt.
    unsigned int loff = (unsigned int)(uintptr_t)(void*)l;
    unsigned long long gaddr = (unsigned long long)(uintptr_t)(const void*)g;
    asm volatile("global_load_async_to_lds_b128 %0, %1, off"
                 :: "v"(loff), "v"(gaddr)
                 : "memory");
#else
    *(v8h*)l = *(const v8h*)g;
#endif
  };

  // --- Q A-tile (16 queries x 32 dh), registers for the whole kernel
  const _Float16* qrow = Q + ((size_t)(b * SEQ + qbase + ln)) * 512 + h * 32;
  v16h qa = pack16(*(const v8h*)(qrow + half * 16),
                   *(const v8h*)(qrow + half * 16 + 8));

  const float scale = 0.044194173824159216f;   // 1/sqrt(512) per reference
  float m[8], l[8];
  v8f acc0 = {}, acc1 = {};
#pragma unroll
  for (int v = 0; v < 8; ++v) { m[v] = -__builtin_inff(); l[v] = 0.f; }

  int ibuf = 0;
  fillKV(0, 0);
  for (int kb = 0; kb < NVIS; kb += 32, ibuf ^= 1) {
#if defined(USE_ASYNC_LDS)
    asm volatile("s_wait_asynccnt 0" ::: "memory");   // own portion landed
#endif
    __syncthreads();                          // panel visible; prev buf free
    if (kb + 32 < NVIS) fillKV(ibuf ^ 1, kb + 32);

    // B tiles of K^T (32dh x 16keys): lane = key column, contiguous dh run
    v16h k0 = *(const v16h*)&stageK[ibuf][ln * 32 + half * 16];
    v16h k1 = *(const v16h*)&stageK[ibuf][(16 + ln) * 32 + half * 16];
    v8f zero = {};
    v8f s0 = __builtin_amdgcn_wmma_f32_16x16x32_f16(false, qa, false, k0, (short)0, zero, false, false);
    v8f s1 = __builtin_amdgcn_wmma_f32_16x16x32_f16(false, qa, false, k1, (short)0, zero, false, false);

    float p0[8], p1[8], cc[8], rs[8], mn[8];
#pragma unroll
    for (int v = 0; v < 8; ++v) {
      float x0 = s0[v] * scale, x1 = s1[v] * scale;
      float bm = fmaxf(x0, x1);               // row stat: reduce over 16-lane half
      bm = fmaxf(bm, __shfl_xor(bm, 1));
      bm = fmaxf(bm, __shfl_xor(bm, 2));
      bm = fmaxf(bm, __shfl_xor(bm, 4));
      bm = fmaxf(bm, __shfl_xor(bm, 8));
      mn[v] = fmaxf(m[v], bm);
      cc[v] = __expf(m[v] - mn[v]);
      p0[v] = __expf(x0 - mn[v]);
      p1[v] = __expf(x1 - mn[v]);
      float r = p0[v] + p1[v];
      r += __shfl_xor(r, 1);
      r += __shfl_xor(r, 2);
      r += __shfl_xor(r, 4);
      r += __shfl_xor(r, 8);
      rs[v] = r;
    }
#pragma unroll
    for (int v = 0; v < 8; ++v) {
      l[v] = l[v] * cc[v] + rs[v];
      m[v] = mn[v];
      acc0[v] *= cc[v];
      acc1[v] *= cc[v];
      // C/D layout -> LDS row-major P[row][key]
      P[(half * 8 + v) * 32 + ln]      = (_Float16)p0[v];
      P[(half * 8 + v) * 32 + 16 + ln] = (_Float16)p1[v];
    }
    asm volatile("s_wait_dscnt 0" ::: "memory");
    // Re-read P in WMMA A layout (16 queries x 32 keys)
    v16h pa = pack16(*(const v8h*)(P + ln * 32 + half * 16),
                     *(const v8h*)(P + ln * 32 + half * 16 + 8));
    asm volatile("" ::: "memory");
    // B tiles of V (32keys x 32dh): lane = dh column, contiguous key run
    v16h v0 = *(const v16h*)&stageV[ibuf][ln * 32 + half * 16];
    v16h v1 = *(const v16h*)&stageV[ibuf][(16 + ln) * 32 + half * 16];
    acc0 = __builtin_amdgcn_wmma_f32_16x16x32_f16(false, pa, false, v0, (short)0, acc0, false, false);
    acc1 = __builtin_amdgcn_wmma_f32_16x16x32_f16(false, pa, false, v1, (short)0, acc1, false, false);
  }

  _Float16* orow = Att + ((size_t)(b * SEQ + qbase)) * 512 + h * 32;
#pragma unroll
  for (int v = 0; v < 8; ++v) {
    float inv = 1.0f / l[v];
    int row = half * 8 + v;
    orow[(size_t)row * 512 + ln]      = (_Float16)(acc0[v] * inv);
    orow[(size_t)row * 512 + 16 + ln] = (_Float16)(acc1[v] * inv);
  }
}

// ---------------------------------------------------------------------------
// Output projection: O[4096][1024] = f32( A[4096][512] @ WoT^T + bo )
// WoT is [1024][512].  One wave -> 16x64 strip.
// ---------------------------------------------------------------------------
__global__ void cma_outproj_kernel(const _Float16* __restrict__ A,
                                   const _Float16* __restrict__ WoT,
                                   const float* __restrict__ bo,
                                   float* __restrict__ O) {
  const int lane = threadIdx.x & 31;
  const int wid  = (blockIdx.x * blockDim.x + threadIdx.x) >> 5;
  const int ln = lane & 15, half = lane >> 4;
  const int nt = wid & 15;     // 1024/64 strips
  const int mt = wid >> 4;
  const int mbase = mt * 16, nbase = nt * 64;

  v8f acc[4] = {};
  const _Float16* arow = A + (size_t)(mbase + ln) * 512;
  for (int kk = 0; kk < 512; kk += 32) {
    v16h a = pack16(*(const v8h*)(arow + kk + half * 16),
                    *(const v8h*)(arow + kk + half * 16 + 8));
#pragma unroll
    for (int t = 0; t < 4; ++t) {
      v16h bm = *(const v16h*)(WoT + (size_t)(nbase + t * 16 + ln) * 512 + kk + half * 16);
      acc[t] = __builtin_amdgcn_wmma_f32_16x16x32_f16(
          false, a, false, bm, (short)0, acc[t], false, false);
    }
  }
#pragma unroll
  for (int t = 0; t < 4; ++t) {
    int col = nbase + t * 16 + ln;
    float bv = bo[col];
#pragma unroll
    for (int v = 0; v < 8; ++v) {
      int row = mbase + half * 8 + v;
      O[(size_t)row * 1024 + col] = acc[t][v] + bv;
    }
  }
}

// ---------------------------------------------------------------------------
extern "C" void kernel_launch(void* const* d_in, const int* in_sizes, int n_in,
                              void* d_out, int out_size, void* d_ws, size_t ws_size,
                              hipStream_t stream) {
  const float* text = (const float*)d_in[0];
  const float* vis  = (const float*)d_in[1];
  const float* Wq   = (const float*)d_in[2];
  const float* bq   = (const float*)d_in[3];
  const float* Wk   = (const float*)d_in[4];
  const float* bk   = (const float*)d_in[5];
  const float* Wv   = (const float*)d_in[6];
  const float* bv   = (const float*)d_in[7];
  const float* Wo   = (const float*)d_in[8];
  const float* bo   = (const float*)d_in[9];
  float* out = (float*)d_out;

  char* ws = (char*)d_ws;
  const size_t MB = 1u << 20;
  _Float16* WqT = (_Float16*)(ws + 0 * MB);   // 512x1024 f16 = 1MB
  _Float16* WkT = (_Float16*)(ws + 1 * MB);
  _Float16* WvT = (_Float16*)(ws + 2 * MB);
  _Float16* WoT = (_Float16*)(ws + 3 * MB);   // 1024x512 f16 = 1MB
  _Float16* Qf  = (_Float16*)(ws + 4 * MB);   // 4096x512 f16 = 4MB
  _Float16* Kf  = (_Float16*)(ws + 8 * MB);
  _Float16* Vr  = (_Float16*)(ws + 12 * MB);
  _Float16* Vt  = (_Float16*)(ws + 16 * MB);
  _Float16* At  = (_Float16*)(ws + 20 * MB);  // total 24MB

  // 1) weights -> transposed f16
  cma_wT_kernel<<<(1024 * 512 + 255) / 256, 256, 0, stream>>>(Wq, WqT, 1024, 512);
  cma_wT_kernel<<<(1024 * 512 + 255) / 256, 256, 0, stream>>>(Wk, WkT, 1024, 512);
  cma_wT_kernel<<<(1024 * 512 + 255) / 256, 256, 0, stream>>>(Wv, WvT, 1024, 512);
  cma_wT_kernel<<<(512 * 1024 + 255) / 256, 256, 0, stream>>>(Wo, WoT, 512, 1024);

  // 2) Q/K/V projections (M = B*S = B*NV = 4096): 2048 waves each -> 256 blocks
  cma_proj_kernel<<<256, 256, 0, stream>>>(text, WqT, bq, Qf);
  cma_proj_kernel<<<256, 256, 0, stream>>>(vis,  WkT, bk, Kf);
  cma_proj_kernel<<<256, 256, 0, stream>>>(vis,  WvT, bv, Vr);

  // 3) V -> [B,H,Dh,NV]
  cma_vT_kernel<<<(1 << 21) / 256, 256, 0, stream>>>(Vr, Vt);

  // 4) flash attention: 512 blocks x 8 waves (block shares (b,h) K/V panels)
  cma_attn_kernel<<<512, 256, 0, stream>>>(Qf, Kf, Vt, At);

  // 5) output projection: 4096 waves -> 512 blocks
  cma_outproj_kernel<<<512, 256, 0, stream>>>(At, WoT, bo, out);
}
